// RCNNModule_23519240913469
// MI455X (gfx1250) — compile-verified, matrix operand order checked by reading, not run on hardware
//
#include <hip/hip_runtime.h>
#include <hip/hip_bf16.h>
#include <stdint.h>

#define P 7
#define C_IN 256
#define N_ROIS 1024
#define D_IN (C_IN * P * P)   // 12544
#define FC1 1024
#define FC2 1024
#define N_CLS 81
#define N_BBOX 324
#define N_HEAD (N_CLS + N_BBOX) // 405

typedef __attribute__((ext_vector_type(16))) __bf16 v16bf;
typedef __attribute__((ext_vector_type(8)))  float  v8f;

union Frag { v16bf v; uint4 u[2]; };

__device__ __forceinline__ unsigned short f32_to_bf16(float f) {
    unsigned int u = __float_as_uint(f);
    unsigned int r = u + 0x7FFFu + ((u >> 16) & 1u); // RNE
    return (unsigned short)(r >> 16);
}

// Async global -> LDS copy (CDNA5 TDM-adjacent path, tracked by ASYNCcnt).
// VDST operand = per-lane LDS byte offset; address in VGPR pair; SADDR=off.
__device__ __forceinline__ void async_copy_b128(unsigned lds_off,
                                                const unsigned short* gptr) {
    asm volatile("global_load_async_to_lds_b128 %0, %1, off"
                 :: "v"(lds_off), "v"(gptr)
                 : "memory");
}
__device__ __forceinline__ void wait_async_le3() {
    asm volatile("s_wait_asynccnt 0x3" ::: "memory");
}
__device__ __forceinline__ void wait_async_0() {
    asm volatile("s_wait_asynccnt 0x0" ::: "memory");
}
__device__ __forceinline__ unsigned lds_offset_of(const void* p) {
    return (unsigned)(uintptr_t)p; // generic LDS addr: [31:0] = LDS offset
}

// ---------------------------------------------------------------------------
// Kernel 1: ROIAlign (aligned=False style), writes bf16 GEMM-ready A matrix.
// One block per roi, 256 threads = one channel each.
// ---------------------------------------------------------------------------
__global__ __launch_bounds__(256) void roi_align_bf16_kernel(
    const float* __restrict__ f0, const float* __restrict__ f1,
    const float* __restrict__ f2, const float* __restrict__ f3,
    const float* __restrict__ rois, unsigned short* __restrict__ Abf)
{
    const int r = blockIdx.x;
    const int t = threadIdx.x;

    __shared__ int   sy0[14], sy1[14], sx0[14], sx1[14];
    __shared__ float sly[14], slx[14];

    const float x1 = rois[r * 5 + 1];
    const float y1 = rois[r * 5 + 2];
    const float x2 = rois[r * 5 + 3];
    const float y2 = rois[r * 5 + 4];

    float area_s = sqrtf((x2 - x1 + 1.0f) * (y2 - y1 + 1.0f));
    int lvl = (int)floorf(log2f(area_s / 56.0f + 1e-6f));
    lvl = min(max(lvl, 0), 3);
    const int H = 256 >> lvl;                  // square feature maps
    const float sc = 1.0f / (float)(4 << lvl); // 1/stride

    if (t < 28) {
        int i = t % 14;
        float step = ((float)i + 0.5f) / 14.0f;
        if (t < 14) {
            float ya = y1 * sc, yb = y2 * sc;
            float rh = fmaxf(yb - ya, 1.0f);
            float yc = fminf(fmaxf(ya + step * rh, 0.0f), (float)(H - 1));
            int i0 = (int)floorf(yc);
            sy0[i] = i0;
            sy1[i] = min(i0 + 1, H - 1);
            sly[i] = yc - (float)i0;
        } else {
            float xa = x1 * sc, xb = x2 * sc;
            float rw = fmaxf(xb - xa, 1.0f);
            float xc = fminf(fmaxf(xa + step * rw, 0.0f), (float)(H - 1));
            int i0 = (int)floorf(xc);
            sx0[i] = i0;
            sx1[i] = min(i0 + 1, H - 1);
            slx[i] = xc - (float)i0;
        }
    }
    __syncthreads();

    const int b = (int)rois[r * 5 + 0];
    const float* feat = (lvl == 0) ? f0 : (lvl == 1) ? f1 : (lvl == 2) ? f2 : f3;
    const float* plane = feat + ((size_t)(b * C_IN + t)) * (size_t)H * (size_t)H;

    unsigned short* outc = Abf + (size_t)r * D_IN + (size_t)t * (P * P);

    for (int py = 0; py < P; ++py) {
        for (int px = 0; px < P; ++px) {
            float s = 0.0f;
            #pragma unroll
            for (int a2 = 0; a2 < 2; ++a2) {
                const int iy = 2 * py + a2;
                const float* row0 = plane + (size_t)sy0[iy] * H;
                const float* row1 = plane + (size_t)sy1[iy] * H;
                const float fy = sly[iy];
                #pragma unroll
                for (int b2 = 0; b2 < 2; ++b2) {
                    const int ix = 2 * px + b2;
                    const int c0 = sx0[ix], c1 = sx1[ix];
                    const float fx = slx[ix];
                    float v00 = row0[c0], v01 = row0[c1];
                    float v10 = row1[c0], v11 = row1[c1];
                    float top = v00 + fx * (v01 - v00);
                    float bot = v10 + fx * (v11 - v10);
                    s += top + fy * (bot - top);
                }
            }
            outc[py * P + px] = f32_to_bf16(s * 0.25f);
        }
    }
}

// ---------------------------------------------------------------------------
// Kernel 2: fp32 -> bf16 conversion (weights)
// ---------------------------------------------------------------------------
__global__ void cvt_f32_bf16_kernel(const float* __restrict__ in,
                                    unsigned short* __restrict__ out, size_t n)
{
    size_t i = (size_t)blockIdx.x * 256u + threadIdx.x;
    size_t stride = (size_t)gridDim.x * 256u;
    for (; i < n; i += stride) out[i] = f32_to_bf16(in[i]);
}

// ---------------------------------------------------------------------------
// Kernel 3: WMMA bf16 GEMM.  C[M,N] = A[M,K] * B[N,K]^T + bias, opt. ReLU.
// Block: 256 thr = 8 waves (4M x 2N), block tile 128x64, BK=32.
// Double-buffered LDS fed by async global->LDS copies (ASYNCcnt pipeline):
// while WMMAs consume buffer `cur`, tile k+32 streams into buffer `cur^1`;
// s_wait_asynccnt<=3 leaves only the next tile's 3 ops in flight.
// LDS rows padded to 80B (16B aligned; conflict-free fragment reads).
// Fragment layout per ISA 7.12.2: lane group g = lane>>4 holds K chunks
// [8g, 8g+8) and [16+8g, 24+8g).
// ---------------------------------------------------------------------------
__global__ __launch_bounds__(256) void gemm_wmma_bf16_kernel(
    const unsigned short* __restrict__ A,   // [M,K] bf16
    const unsigned short* __restrict__ B,   // [N,K] bf16 (row n contiguous K)
    const float* __restrict__ bias1, const float* __restrict__ bias2, int nsplit,
    int M, int N, int K,
    float* __restrict__ outF, int ldF,
    unsigned short* __restrict__ outB, int ldB,
    int doRelu)
{
    __shared__ unsigned short lsA[2][128 * 40]; // 80B pitch, double buffered
    __shared__ unsigned short lsB[2][64 * 40];

    const int t     = threadIdx.x;
    const int lane  = t & 31;
    const int wave  = t >> 5;
    const int warpM = wave & 3;
    const int warpN = wave >> 2;
    const int m0 = blockIdx.y * 128;
    const int n0 = blockIdx.x * 64;
    const int g    = (lane >> 4) & 1;
    const int rsel = lane & 15;

    const int arow = t >> 1, ahalf = t & 1;  // A stage: 32B per thread
    const int brow = t >> 2, bq = t & 3;     // B stage: 16B per thread

    // B rows past N are clamped (not zeroed): junk only reaches accumulator
    // lanes whose output column >= N, which the epilogue never stores.
    const int bn = min(n0 + brow, N - 1);
    const unsigned short* aSrc = A + (size_t)(m0 + arow) * K + ahalf * 16;
    const unsigned short* bSrc = B + (size_t)bn * K + bq * 8;

    unsigned aDst[2], bDst[2];
    #pragma unroll
    for (int buf = 0; buf < 2; ++buf) {
        aDst[buf] = lds_offset_of(&lsA[buf][arow * 40 + ahalf * 16]);
        bDst[buf] = lds_offset_of(&lsB[buf][brow * 40 + bq * 8]);
    }

    v8f acc[2][2] = {};

    // prologue: stream tile 0
    async_copy_b128(aDst[0],      aSrc + 0);
    async_copy_b128(aDst[0] + 16, aSrc + 8);
    async_copy_b128(bDst[0],      bSrc + 0);

    for (int k0 = 0; k0 < K; k0 += 32) {
        const int  cur  = (k0 >> 5) & 1;
        const bool more = (k0 + 32) < K;

        if (more) { // stream next tile into the other buffer
            const int nxt = cur ^ 1;
            async_copy_b128(aDst[nxt],      aSrc + (k0 + 32));
            async_copy_b128(aDst[nxt] + 16, aSrc + (k0 + 40));
            async_copy_b128(bDst[nxt],      bSrc + (k0 + 32));
            wait_async_le3();   // current tile's 3 ops retired
        } else {
            wait_async_0();
        }
        __syncthreads();        // all waves' async data visible in LDS

        Frag a[2], b[2];
        #pragma unroll
        for (int mi = 0; mi < 2; ++mi) {
            const unsigned short* rp = &lsA[cur][(warpM * 32 + mi * 16 + rsel) * 40];
            a[mi].u[0] = *(const uint4*)(rp + 8 * g);
            a[mi].u[1] = *(const uint4*)(rp + 16 + 8 * g);
        }
        #pragma unroll
        for (int ni = 0; ni < 2; ++ni) {
            const unsigned short* rp = &lsB[cur][(warpN * 32 + ni * 16 + rsel) * 40];
            b[ni].u[0] = *(const uint4*)(rp + 8 * g);
            b[ni].u[1] = *(const uint4*)(rp + 16 + 8 * g);
        }

        #pragma unroll
        for (int mi = 0; mi < 2; ++mi)
            #pragma unroll
            for (int ni = 0; ni < 2; ++ni)
                acc[mi][ni] = __builtin_amdgcn_wmma_f32_16x16x32_bf16(
                    false, a[mi].v, false, b[ni].v,
                    (short)0, acc[mi][ni], false, false);

        __syncthreads();        // buffer `cur` free for the iter-after-next
    }

    // ---- epilogue: bias (+split), relu, f32 and/or bf16 store ----
    #pragma unroll
    for (int mi = 0; mi < 2; ++mi) {
        #pragma unroll
        for (int ni = 0; ni < 2; ++ni) {
            const int col = n0 + warpN * 32 + ni * 16 + rsel;
            if (col >= N) continue;
            const float bv = (col < nsplit) ? bias1[col] : bias2[col - nsplit];
            #pragma unroll
            for (int j = 0; j < 8; ++j) {
                const int row = m0 + warpM * 32 + mi * 16 + j + 8 * (lane >> 4);
                float v = acc[mi][ni][j] + bv;
                if (doRelu) v = fmaxf(v, 0.0f);
                if (outF) outF[(size_t)row * ldF + col] = v;
                if (outB) outB[(size_t)row * ldB + col] = f32_to_bf16(v);
            }
        }
    }
}

// ---------------------------------------------------------------------------
// Kernel 4: softmax over 81 classes + assemble (rois, cls_prob, bbox_pred).
// One block (128 thr) per roi.
// ---------------------------------------------------------------------------
__global__ __launch_bounds__(128) void softmax_assemble_kernel(
    const float* __restrict__ s3,   // [N_ROIS, N_HEAD]
    const float* __restrict__ rois, // [N_ROIS, 5]
    float* __restrict__ out)
{
    const int r = blockIdx.x;
    const int t = threadIdx.x;
    __shared__ float buf[128];

    const float* srow = s3 + (size_t)r * N_HEAD;

    float v = (t < N_CLS) ? srow[t] : -3.4e38f;
    buf[t] = v;
    __syncthreads();
    for (int s = 64; s > 0; s >>= 1) {
        if (t < s) buf[t] = fmaxf(buf[t], buf[t + s]);
        __syncthreads();
    }
    const float mx = buf[0];
    __syncthreads();

    float e = (t < N_CLS) ? __expf(v - mx) : 0.0f;
    buf[t] = e;
    __syncthreads();
    for (int s = 64; s > 0; s >>= 1) {
        if (t < s) buf[t] += buf[t + s];
        __syncthreads();
    }
    const float inv = 1.0f / buf[0];

    const size_t OFS_CLS  = (size_t)N_ROIS * 5;
    const size_t OFS_BBOX = OFS_CLS + (size_t)N_ROIS * N_CLS;

    if (t < 5)      out[(size_t)r * 5 + t] = rois[(size_t)r * 5 + t];
    if (t < N_CLS)  out[OFS_CLS + (size_t)r * N_CLS + t] = e * inv;
    for (int i = t; i < N_BBOX; i += 128)
        out[OFS_BBOX + (size_t)r * N_BBOX + i] = srow[N_CLS + i];
}

// ---------------------------------------------------------------------------
extern "C" void kernel_launch(void* const* d_in, const int* in_sizes, int n_in,
                              void* d_out, int out_size, void* d_ws, size_t ws_size,
                              hipStream_t stream) {
    const float* feat0  = (const float*)d_in[0];
    const float* feat1  = (const float*)d_in[1];
    const float* feat2  = (const float*)d_in[2];
    const float* feat3  = (const float*)d_in[3];
    const float* rois   = (const float*)d_in[4];
    const float* fc1_w  = (const float*)d_in[5];
    const float* fc1_b  = (const float*)d_in[6];
    const float* fc2_w  = (const float*)d_in[7];
    const float* fc2_b  = (const float*)d_in[8];
    const float* cls_w  = (const float*)d_in[9];
    const float* cls_b  = (const float*)d_in[10];
    const float* bbox_w = (const float*)d_in[11];
    const float* bbox_b = (const float*)d_in[12];
    float* out = (float*)d_out;

    char* ws = (char*)d_ws;
    size_t o = 0;
    auto take = [&](size_t bytes) -> char* {
        char* p = ws + o;
        o += (bytes + 255) & ~(size_t)255;
        return p;
    };
    unsigned short* Abf  = (unsigned short*)take((size_t)N_ROIS * D_IN * 2);
    unsigned short* W1bf = (unsigned short*)take((size_t)FC1 * D_IN * 2);
    unsigned short* W2bf = (unsigned short*)take((size_t)FC2 * FC1 * 2);
    unsigned short* W3bf = (unsigned short*)take((size_t)N_HEAD * FC2 * 2);
    unsigned short* X1bf = (unsigned short*)take((size_t)N_ROIS * FC1 * 2);
    unsigned short* X2bf = (unsigned short*)take((size_t)N_ROIS * FC2 * 2);
    float*          S3   = (float*)take((size_t)N_ROIS * N_HEAD * 4);

    // 1) ROIAlign -> bf16 activation matrix A [1024, 12544]
    roi_align_bf16_kernel<<<N_ROIS, 256, 0, stream>>>(feat0, feat1, feat2, feat3,
                                                      rois, Abf);

    // 2) weight conversions (cls_w ++ bbox_w concatenated into the head matrix)
    {
        size_t n1 = (size_t)FC1 * D_IN;
        cvt_f32_bf16_kernel<<<(unsigned)((n1 + 255) / 256), 256, 0, stream>>>(fc1_w, W1bf, n1);
        size_t n2 = (size_t)FC2 * FC1;
        cvt_f32_bf16_kernel<<<(unsigned)((n2 + 255) / 256), 256, 0, stream>>>(fc2_w, W2bf, n2);
        size_t n3 = (size_t)N_CLS * FC2;
        cvt_f32_bf16_kernel<<<(unsigned)((n3 + 255) / 256), 256, 0, stream>>>(cls_w, W3bf, n3);
        size_t n4 = (size_t)N_BBOX * FC2;
        cvt_f32_bf16_kernel<<<(unsigned)((n4 + 255) / 256), 256, 0, stream>>>(
            bbox_w, W3bf + (size_t)N_CLS * FC2, n4);
    }

    // 3) fc1: [1024,12544] x [1024,12544]^T + b, ReLU -> bf16 X1
    gemm_wmma_bf16_kernel<<<dim3(FC1 / 64, N_ROIS / 128), 256, 0, stream>>>(
        Abf, W1bf, fc1_b, fc1_b, FC1,
        N_ROIS, FC1, D_IN,
        nullptr, 0, X1bf, FC1, 1);

    // 4) fc2: [1024,1024] x [1024,1024]^T + b, ReLU -> bf16 X2
    gemm_wmma_bf16_kernel<<<dim3(FC2 / 64, N_ROIS / 128), 256, 0, stream>>>(
        X1bf, W2bf, fc2_b, fc2_b, FC2,
        N_ROIS, FC2, FC1,
        nullptr, 0, X2bf, FC2, 1);

    // 5) heads: [1024,1024] x [405,1024]^T + (cls_b ++ bbox_b) -> f32 S3
    gemm_wmma_bf16_kernel<<<dim3((N_HEAD + 63) / 64, N_ROIS / 128), 256, 0, stream>>>(
        X2bf, W3bf, cls_b, bbox_b, N_CLS,
        N_ROIS, N_HEAD, FC2,
        S3, N_HEAD, nullptr, 0, 0);

    // 6) softmax + assemble output tuple
    softmax_assemble_kernel<<<N_ROIS, 128, 0, stream>>>(S3, rois, out);
}
